// KabaddiAFGN_19851338842752
// MI455X (gfx1250) — compile-verified, matrix operand order checked by reading, not run on hardware
//
#include <hip/hip_runtime.h>
#include <hip/hip_bf16.h>
#include <cstdint>
#include <cstddef>

// ---------------------------------------------------------------------------
// MI455X / gfx1250 implementation. All heavy GEMMs use v_wmma_f32_16x16x32_f16
// (f16 inputs, f32 accumulate); each wave computes a 32x16 tile (two WMMAs
// back-to-back sharing the B fragment). Streamed activations are f16 to halve
// HBM traffic (the T*E*H edge tensor alone is 268MB in f32). GRU recurrence
// keeps hidden state resident in LDS and does h@Whh^T with WMMA. Attention
// k/v staging uses GLOBAL_LOAD_ASYNC_TO_LDS (ASYNCcnt-tracked DMA).
// ---------------------------------------------------------------------------

typedef __attribute__((ext_vector_type(16))) _Float16 v16h;
typedef __attribute__((ext_vector_type(8)))  float    v8f;
typedef __attribute__((ext_vector_type(4)))  int      int4v;
typedef _Float16 h16_t;

// Pointer types matching the async-to-LDS builtin signature (int4 pointee,
// AS1 global source / AS3 LDS destination).
typedef __attribute__((address_space(1))) int4v* gas_i4p;
typedef __attribute__((address_space(3))) int4v* las_i4p;

#define TT   32
#define BB   256
#define MXN  8
#define HH   128
#define NN   2048      /* BB*MXN nodes per timestep   */
#define EE   16384     /* BB*MXN*MXN edges / timestep */
#define TN   65536     /* TT*NN                       */
#define TE   524288    /* TT*EE                       */
#define TB   8192      /* TT*BB                       */
#define G3   384       /* 3*HH                        */

__device__ __forceinline__ float geluf(float x) {
  return 0.5f * x * (1.0f + erff(x * 0.70710678118654752f));
}
__device__ __forceinline__ float sigf(float x) {
  return 1.0f / (1.0f + __expf(-x));
}

// Load one WMMA f16 fragment half per lane: 8 contiguous halves at p (K block
// [k0+half*8, +8)) and 8 at p+16 (K block [k0+16+half*8, +8)). Matches the
// documented 16-bit A/B VGPR layout for 16x16x32.
__device__ __forceinline__ v16h load_frag16(const h16_t* p) {
  union { int4 q[2]; v16h v; } u;
  u.q[0] = *reinterpret_cast<const int4*>(p);
  u.q[1] = *reinterpret_cast<const int4*>(p + 16);
  return u.v;
}

__device__ __forceinline__ void async_ld_b128(const h16_t* g, h16_t* l) {
#if __has_builtin(__builtin_amdgcn_global_load_async_to_lds_b128)
  __builtin_amdgcn_global_load_async_to_lds_b128(
      (gas_i4p)(uintptr_t)g, (las_i4p)(uint32_t)(uintptr_t)l, 0, 0);
#endif
}
__device__ __forceinline__ void wait_async0() {
#if __has_builtin(__builtin_amdgcn_s_wait_asynccnt)
  __builtin_amdgcn_s_wait_asynccnt(0);
#else
  asm volatile("s_wait_asynccnt 0x0" ::: "memory");
#endif
}

// --------------------------- f32 -> f16 cast -------------------------------
__global__ void cast_f16_kernel(const float* __restrict__ in,
                                h16_t* __restrict__ out, int n) {
  int i = blockIdx.x * 256 + threadIdx.x;
  if (i < n) out[i] = (h16_t)in[i];
}

// ------------- first-layer lin (tiny K) + gelu (+ optional LN) -------------
__global__ __launch_bounds__(256)
void lin_small_kernel(const float* __restrict__ X, int Kin,
                      const float* __restrict__ W, const float* __restrict__ bv,
                      int doLN, const float* __restrict__ g,
                      const float* __restrict__ be,
                      h16_t* __restrict__ out, int M) {
  __shared__ float red[2][128];
  int r2  = threadIdx.x >> 7;
  int j   = threadIdx.x & 127;
  int row = blockIdx.x * 2 + r2;
  float acc = bv[j];
  if (row < M) {
    const float* xr = X + (size_t)row * Kin;
    const float* wr = W + (size_t)j * Kin;
    for (int k = 0; k < Kin; ++k) acc += xr[k] * wr[k];
  }
  float h = geluf(acc);
  if (doLN) {
    red[r2][j] = h; __syncthreads();
    for (int s = 64; s > 0; s >>= 1) {
      if (j < s) red[r2][j] += red[r2][j + s];
      __syncthreads();
    }
    float m = red[r2][0] * (1.0f / 128.0f);
    __syncthreads();
    float d = h - m;
    red[r2][j] = d * d; __syncthreads();
    for (int s = 64; s > 0; s >>= 1) {
      if (j < s) red[r2][j] += red[r2][j + s];
      __syncthreads();
    }
    float var = red[r2][0] * (1.0f / 128.0f);
    h = d * rsqrtf(var + 1e-5f) * g[j] + be[j];
  }
  if (row < M) out[(size_t)row * HH + j] = (h16_t)h;
}

// --------------------------- generic WMMA GEMM -----------------------------
// out = act(A16[M,K] @ W16[N,K]^T + bias). Each wave computes a 32x16 tile:
// two accumulators share one B fragment -> 2 independent WMMAs per K step.
__global__ __launch_bounds__(256)
void gemm16_kernel(const h16_t* __restrict__ A, const h16_t* __restrict__ W,
                   const float* __restrict__ bias,
                   float* __restrict__ outF, h16_t* __restrict__ outH,
                   int M, int Nout, int K, int act) {
  int w    = threadIdx.x >> 5;
  int lane = threadIdx.x & 31;
  int hl   = lane >> 4;
  int mn   = lane & 15;
  size_t tileM = (size_t)blockIdx.x * 32;
  const h16_t* arow0 = A + (tileM + mn) * (size_t)K;
  const h16_t* arow1 = A + (tileM + 16 + mn) * (size_t)K;
  for (int nt = blockIdx.y * 8 + w; nt * 16 < Nout; nt += gridDim.y * 8) {
    int tileN = nt * 16;
    const h16_t* brow = W + (size_t)(tileN + mn) * K;
    v8f c0 = {0.f, 0.f, 0.f, 0.f, 0.f, 0.f, 0.f, 0.f};
    v8f c1 = {0.f, 0.f, 0.f, 0.f, 0.f, 0.f, 0.f, 0.f};
    for (int k0 = 0; k0 < K; k0 += 32) {
      if (k0 + 32 < K) {  // pull next K slice into near caches (WGP scope)
        __builtin_prefetch(arow0 + k0 + 32, 0, 3);
        __builtin_prefetch(arow1 + k0 + 32, 0, 3);
        __builtin_prefetch(brow + k0 + 32, 0, 3);
      }
      v16h a0 = load_frag16(arow0 + k0 + hl * 8);
      v16h a1 = load_frag16(arow1 + k0 + hl * 8);
      v16h b  = load_frag16(brow + k0 + hl * 8);
      c0 = __builtin_amdgcn_wmma_f32_16x16x32_f16(false, a0, false, b,
                                                  (short)0, c0, false, false);
      c1 = __builtin_amdgcn_wmma_f32_16x16x32_f16(false, a1, false, b,
                                                  (short)0, c1, false, false);
    }
#pragma unroll
    for (int r = 0; r < 8; ++r) {
      int    col  = tileN + mn;
      float  bb   = bias ? bias[col] : 0.f;
      size_t row0 = tileM + (size_t)hl * 8 + r;
      size_t row1 = row0 + 16;
      float  v0   = c0[r] + bb;
      float  v1   = c1[r] + bb;
      if (act == 1) { v0 = geluf(v0); v1 = geluf(v1); }
      if (outF) {
        outF[row0 * (size_t)Nout + col] = v0;
        outF[row1 * (size_t)Nout + col] = v1;
      }
      if (outH) {
        outH[row0 * (size_t)Nout + col] = (h16_t)v0;
        outH[row1 * (size_t)Nout + col] = (h16_t)v1;
      }
    }
  }
}

// ------------------ transformer-conv attention (fused) ---------------------
// One wave per (t, graph). lane = dst*4 + head. Exact 8-wide segment softmax.
// Adds messages into xi (pre-loaded with x@Ws^T + bs). k/v staged into LDS via
// async DMA (ASYNCcnt) when the toolchain exposes the gfx1250 builtin.
__global__ __launch_bounds__(256)
void tconv_attn_kernel(const h16_t* __restrict__ q16,
                       const h16_t* __restrict__ k16,
                       const h16_t* __restrict__ v16,
                       const h16_t* __restrict__ e16,
                       float* __restrict__ xi, int ngt) {
  __shared__ __align__(16) h16_t sk[8][8 * 128];
  __shared__ __align__(16) h16_t sv[8][8 * 128];
  int w    = threadIdx.x >> 5;
  int lane = threadIdx.x & 31;
  int gt   = blockIdx.x * 8 + w;
  bool act = gt < ngt;
  int t  = act ? (gt / BB) : 0;
  int gr = act ? (gt % BB) : 0;
  size_t nodebase = (size_t)t * NN + (size_t)gr * MXN;

#if __has_builtin(__builtin_amdgcn_global_load_async_to_lds_b128)
  if (act) {
    const h16_t* kg = k16 + nodebase * HH;
    const h16_t* vg = v16 + nodebase * HH;
#pragma unroll
    for (int c = 0; c < 4; ++c) {
      int idx = c * 256 + lane * 8;  // halves; 16B per lane per op
      async_ld_b128(kg + idx, &sk[w][idx]);
      async_ld_b128(vg + idx, &sv[w][idx]);
    }
  }
  wait_async0();
  if (!act) return;
#else
  if (act) {
    for (int i = lane; i < MXN * HH; i += 32) {
      sk[w][i] = k16[nodebase * HH + i];
      sv[w][i] = v16[nodebase * HH + i];
    }
  }
  __syncthreads();
  if (!act) return;
#endif

  int d  = lane >> 2;          // destination node (0..7)
  int hd = lane & 3;           // head (0..3)
  int cb = hd * 32;            // column base within H
  size_t edgebase = (size_t)t * EE + (size_t)gr * (MXN * MXN);

  float qv[32];
  const h16_t* qp = q16 + (nodebase + d) * HH + cb;
#pragma unroll
  for (int j = 0; j < 32; ++j) qv[j] = (float)qp[j];

  float al[8];
  for (int s = 0; s < 8; ++s) {
    const h16_t* ep = e16 + (edgebase + (size_t)s * MXN + d) * HH + cb;
    float acc = 0.f;
#pragma unroll
    for (int j = 0; j < 32; ++j)
      acc += qv[j] * ((float)sk[w][s * HH + cb + j] + (float)ep[j]);
    al[s] = acc * 0.17677669529663688f;  // 1/sqrt(DH)
  }
  float mx = al[0];
  for (int s = 1; s < 8; ++s) mx = fmaxf(mx, al[s]);
  float den = 0.f;
  for (int s = 0; s < 8; ++s) { al[s] = __expf(al[s] - mx); den += al[s]; }
  float inv = 1.0f / (den + 1e-16f);

  float outv[32];
#pragma unroll
  for (int j = 0; j < 32; ++j) outv[j] = 0.f;
  for (int s = 0; s < 8; ++s) {
    float a = al[s] * inv;
    const h16_t* ep = e16 + (edgebase + (size_t)s * MXN + d) * HH + cb;
#pragma unroll
    for (int j = 0; j < 32; ++j)
      outv[j] += a * ((float)sv[w][s * HH + cb + j] + (float)ep[j]);
  }
  float* op = xi + (nodebase + d) * HH + cb;
#pragma unroll
  for (int j = 0; j < 32; ++j) op[j] += outv[j];
}

// ------------------- xf = LN(xf + gelu(xi)) * g + b ------------------------
__global__ __launch_bounds__(256)
void ln_residual_kernel(const float* __restrict__ xin,
                        const float* __restrict__ xi,
                        const float* __restrict__ g, const float* __restrict__ b,
                        float* __restrict__ xoutF, h16_t* __restrict__ xoutH,
                        int M) {
  __shared__ float red[2][128];
  int r2  = threadIdx.x >> 7;
  int j   = threadIdx.x & 127;
  int row = blockIdx.x * 2 + r2;
  float h = 0.f;
  if (row < M) {
    size_t i = (size_t)row * HH + j;
    h = xin[i] + geluf(xi[i]);
  }
  red[r2][j] = h; __syncthreads();
  for (int s = 64; s > 0; s >>= 1) {
    if (j < s) red[r2][j] += red[r2][j + s];
    __syncthreads();
  }
  float m = red[r2][0] * (1.0f / 128.0f);
  __syncthreads();
  float d = h - m;
  red[r2][j] = d * d; __syncthreads();
  for (int s = 64; s > 0; s >>= 1) {
    if (j < s) red[r2][j] += red[r2][j + s];
    __syncthreads();
  }
  float var = red[r2][0] * (1.0f / 128.0f);
  float o = d * rsqrtf(var + 1e-5f) * g[j] + b[j];
  if (row < M) {
    size_t i = (size_t)row * HH + j;
    xoutF[i] = o;
    xoutH[i] = (h16_t)o;
  }
}

// ------------- gate with global features + (T,N)->(N,T) transpose ----------
__global__ void gate_kernel(const float* __restrict__ xf,
                            const float* __restrict__ uh,
                            h16_t* __restrict__ xseq16) {
  int i = blockIdx.x * 256 + threadIdx.x;
  if (i >= TN * HH) return;
  int col  = i & 127;
  int rown = i >> 7;          // (t, n)
  int t    = rown >> 11;      // N = 2048
  int n    = rown & 2047;
  int gr   = n >> 3;
  float ub = uh[((size_t)t * BB + gr) * HH + col];
  float gv = sigf(ub);
  float v  = xf[(size_t)i] * gv + ub * (1.0f - gv);
  xseq16[(((size_t)n * TT) + t) * HH + col] = (h16_t)v;
}

// ------------------------------ GRU scan -----------------------------------
// Block owns 16 sequences; loops t = 0..T-1. gh = h@Whh^T via WMMA (h in LDS,
// f16 for the matrix unit + f32 master copy), gate math fused, gi precomputed.
__global__ __launch_bounds__(256)
void gru_scan_kernel(const float* __restrict__ gi,
                     const h16_t* __restrict__ Whh,
                     const float* __restrict__ bhh,
                     h16_t* __restrict__ ys, int T) {
  __shared__ __align__(16) h16_t h16s[16 * 128];
  __shared__ float h32s[16 * 128];
  __shared__ float ghs[16 * 384];
  int tid  = threadIdx.x;
  int row0 = blockIdx.x * 16;
  for (int i = tid; i < 16 * 128; i += 256) { h32s[i] = 0.f; h16s[i] = (h16_t)0.f; }
  __syncthreads();

  int w = tid >> 5, lane = tid & 31, hl = lane >> 4, mn = lane & 15;
  int r = tid & 15, cb = (tid >> 4) * 8;

  for (int t = 0; t < T; ++t) {
    // gh = h @ Whh^T  (16 x 384), 24 tiles over 8 waves
    for (int nt = w; nt < 24; nt += 8) {
      int tileN = nt * 16;
      v8f c = {0.f, 0.f, 0.f, 0.f, 0.f, 0.f, 0.f, 0.f};
      for (int k0 = 0; k0 < 128; k0 += 32) {
        v16h a = load_frag16(&h16s[mn * 128 + k0 + hl * 8]);
        v16h b = load_frag16(Whh + (size_t)(tileN + mn) * 128 + k0 + hl * 8);
        c = __builtin_amdgcn_wmma_f32_16x16x32_f16(false, a, false, b,
                                                   (short)0, c, false, false);
      }
#pragma unroll
      for (int rr = 0; rr < 8; ++rr)
        ghs[(hl * 8 + rr) * 384 + tileN + mn] = c[rr];
    }
    __syncthreads();

    size_t gbase = ((size_t)(row0 + r) * T + t) * G3;
#pragma unroll
    for (int jj = 0; jj < 8; ++jj) {
      int j = cb + jj;
      float i_r = gi[gbase + j];
      float i_z = gi[gbase + 128 + j];
      float i_n = gi[gbase + 256 + j];
      float h_r = ghs[r * 384 + j]       + bhh[j];
      float h_z = ghs[r * 384 + 128 + j] + bhh[128 + j];
      float h_n = ghs[r * 384 + 256 + j] + bhh[256 + j];
      float rg = sigf(i_r + h_r);
      float zg = sigf(i_z + h_z);
      float ng = tanhf(i_n + rg * h_n);
      float hn = (1.0f - zg) * ng + zg * h32s[r * 128 + j];
      h32s[r * 128 + j] = hn;
      h16s[r * 128 + j] = (h16_t)hn;
      ys[((size_t)(row0 + r) * T + t) * HH + j] = (h16_t)hn;
    }
    __syncthreads();
  }
}

// ------------ node-max pooling + build global-GRU input (u ++ max) ---------
__global__ void nodemax_kernel(const h16_t* __restrict__ ys2,
                               const h16_t* __restrict__ uh16,
                               h16_t* __restrict__ gin16) {
  int i = blockIdx.x * 256 + threadIdx.x;
  if (i >= BB * TT * HH) return;
  int col = i & 127;
  int bt  = i >> 7;
  int b   = bt >> 5;   // T = 32
  int t   = bt & 31;
  float m = -1e30f;
  for (int n = 0; n < MXN; ++n) {
    float v = (float)ys2[(((size_t)(b * MXN + n)) * TT + t) * HH + col];
    m = fmaxf(m, v);
  }
  size_t base = ((size_t)b * TT + t) * 256;
  gin16[base + col]        = uh16[((size_t)t * BB + b) * HH + col];
  gin16[base + 128 + col]  = (h16_t)m;
}

// -------------- build concat(node_final, node_first) -----------------------
__global__ void ncbuild_kernel(const h16_t* __restrict__ ys2,
                               const h16_t* __restrict__ xseq16,
                               h16_t* __restrict__ nc16) {
  int i = blockIdx.x * 256 + threadIdx.x;
  if (i >= NN * HH) return;
  int col = i & 127;
  int n   = i >> 7;
  nc16[(size_t)n * 256 + col]       = ys2[(((size_t)n) * TT + (TT - 1)) * HH + col];
  nc16[(size_t)n * 256 + 128 + col] = xseq16[((size_t)n * TT + 0) * HH + col];
}

// ------------------ contact head final lin + sigmoid -----------------------
__global__ void contact_final_kernel(const h16_t* __restrict__ h2,
                                     const float* __restrict__ W3,
                                     const float* __restrict__ b3,
                                     float* __restrict__ out) {
  int n = blockIdx.x * 256 + threadIdx.x;
  if (n >= NN) return;
  float acc = b3[0];
  for (int j = 0; j < 64; ++j) acc += (float)h2[(size_t)n * 64 + j] * W3[j];
  int b = n >> 3, idx = n & 7;
  out[b * 12 + idx] = sigf(acc);   // valid mask is all-true for this topology
}

// ---- attention pooling over 8 nodes + build concat(u_final, contact_agg) --
__global__ __launch_bounds__(128)
void attnpool_kernel(const h16_t* __restrict__ ys2,
                     const float* __restrict__ Wp, const float* __restrict__ bp,
                     const h16_t* __restrict__ gys2,
                     h16_t* __restrict__ gc16) {
  __shared__ float red[128];
  __shared__ float saw[8];
  int b = blockIdx.x, tid = threadIdx.x;
  for (int n = 0; n < MXN; ++n) {
    float v = (float)ys2[(((size_t)(b * MXN + n)) * TT + (TT - 1)) * HH + tid] * Wp[tid];
    red[tid] = v; __syncthreads();
    for (int s = 64; s > 0; s >>= 1) {
      if (tid < s) red[tid] += red[tid + s];
      __syncthreads();
    }
    if (tid == 0) saw[n] = red[0] + bp[0];
    __syncthreads();
  }
  if (tid == 0) {
    float mx = saw[0];
    for (int n = 1; n < MXN; ++n) mx = fmaxf(mx, saw[n]);
    float den = 0.f;
    for (int n = 0; n < MXN; ++n) { saw[n] = __expf(saw[n] - mx); den += saw[n]; }
    for (int n = 0; n < MXN; ++n) saw[n] /= den;
  }
  __syncthreads();
  float agg = 0.f;
  for (int n = 0; n < MXN; ++n)
    agg += saw[n] * (float)ys2[(((size_t)(b * MXN + n)) * TT + (TT - 1)) * HH + tid];
  gc16[(size_t)b * 256 + tid]       = gys2[(((size_t)b) * TT + (TT - 1)) * HH + tid];
  gc16[(size_t)b * 256 + 128 + tid] = (h16_t)agg;
}

// ------------------- global head final lin + sigmoid -----------------------
__global__ void global_final_kernel(const h16_t* __restrict__ g2,
                                    const float* __restrict__ W3,
                                    const float* __restrict__ b3,
                                    float* __restrict__ out) {
  int i = blockIdx.x * 256 + threadIdx.x;
  if (i >= BB * 4) return;
  int b = i >> 2, o = i & 3;
  float acc = b3[o];
  for (int j = 0; j < 64; ++j) acc += (float)g2[(size_t)b * 64 + j] * W3[o * 64 + j];
  out[b * 12 + 8 + o] = sigf(acc);
}

// ===========================================================================
// Host orchestration
// ===========================================================================
// Param flattening follows JAX pytree order (dict keys sorted, lists in order):
//  0:x 1:edge_index 2:edge_attr 3:u 4:batch 5:node_idx
//  6:attn.W 7:attn.b | 8..13: contact W1,W2,W3,b1,b2,b3
//  14..23 conv1: We,Wk,Wq,Ws,Wv,be,bk,bq,bs,bv | 24..33 conv2 | 34..43 conv3
//  44..47 edge_mlp: W1,W2,b1,b2
//  48..55 global_gru: l0 Whh,Wih,bhh,bih ; l1 Whh,Wih,bhh,bih
//  56..61 global_head: W1,W2,W3,b1,b2,b3 | 62..65 global_mlp: W1,W2,b1,b2
//  66..73 node_gru: l0 Whh,Wih,bhh,bih ; l1 Whh,Wih,bhh,bih
//  74..79 node_mlp: W1,W2,b1,b2,be,g
//  80..85 norm1 b,g | norm2 b,g | norm3 b,g

extern "C" void kernel_launch(void* const* d_in, const int* in_sizes, int n_in,
                              void* d_out, int out_size, void* d_ws, size_t ws_size,
                              hipStream_t stream) {
  (void)in_sizes; (void)n_in; (void)out_size; (void)ws_size;
  const float* x_in  = (const float*)d_in[0];
  const float* ea_in = (const float*)d_in[2];
  const float* u_in  = (const float*)d_in[3];
  auto P = [&](int i) { return (const float*)d_in[i]; };

  uint8_t* ws = (uint8_t*)d_ws;
  size_t off = 0;
  auto alloc = [&](size_t bytes) -> void* {
    void* p = ws + off;
    off = (off + bytes + 255) & ~(size_t)255;
    return p;
  };

  float* xf    = (float*)alloc((size_t)TN * HH * 4);
  h16_t* xf16  = (h16_t*)alloc((size_t)TN * HH * 2);
  h16_t* q16d  = (h16_t*)alloc((size_t)TN * HH * 2);   // also node lin1 tmp
  h16_t* k16d  = (h16_t*)alloc((size_t)TN * HH * 2);
  h16_t* v16d  = (h16_t*)alloc((size_t)TN * HH * 2);
  float* xi    = (float*)alloc((size_t)TN * HH * 4);
  h16_t* ea16  = (h16_t*)alloc((size_t)TE * HH * 2);
  h16_t* ebuf  = (h16_t*)alloc((size_t)TE * HH * 2);   // edge tmp / e / gi
  float* uhf   = (float*)alloc((size_t)TB * HH * 4);
  h16_t* uh16  = (h16_t*)alloc((size_t)TB * HH * 2);
  h16_t* tmpu  = (h16_t*)alloc((size_t)TB * HH * 2);
  h16_t* xseq  = (h16_t*)alloc((size_t)TN * HH * 2);
  h16_t* ys1   = (h16_t*)alloc((size_t)TN * HH * 2);
  h16_t* ys2   = (h16_t*)alloc((size_t)TN * HH * 2);
  h16_t* gin16 = (h16_t*)alloc((size_t)TB * 256 * 2);
  h16_t* gys1  = (h16_t*)alloc((size_t)TB * HH * 2);
  h16_t* gys2  = (h16_t*)alloc((size_t)TB * HH * 2);
  h16_t* nc16  = (h16_t*)alloc((size_t)NN * 256 * 2);
  h16_t* ch1   = (h16_t*)alloc((size_t)NN * HH * 2);
  h16_t* ch2   = (h16_t*)alloc((size_t)NN * 64 * 2);
  h16_t* gc16  = (h16_t*)alloc((size_t)BB * 256 * 2);
  h16_t* gg1   = (h16_t*)alloc((size_t)BB * HH * 2);
  h16_t* gg2   = (h16_t*)alloc((size_t)BB * 64 * 2);
  float* gi    = (float*)ebuf;  // 65536*384*4 = 100.7MB <= 134.2MB (reused after convs)

  auto castw = [&](int pidx, int n) -> h16_t* {
    h16_t* dst = (h16_t*)alloc((size_t)n * 2);
    cast_f16_kernel<<<(n + 255) / 256, 256, 0, stream>>>(P(pidx), dst, n);
    return dst;
  };

  h16_t* nW2 = castw(75, HH * HH);
  h16_t* eW2 = castw(45, HH * HH);
  h16_t* gW2 = castw(63, HH * HH);
  h16_t* cW[3][5];
  for (int i = 0; i < 3; ++i) {
    int ci = 14 + i * 10;
    for (int k = 0; k < 5; ++k) cW[i][k] = castw(ci + k, HH * HH);  // We,Wk,Wq,Ws,Wv
  }
  h16_t* nWhh0 = castw(66, G3 * HH);
  h16_t* nWih0 = castw(67, G3 * HH);
  h16_t* nWhh1 = castw(70, G3 * HH);
  h16_t* nWih1 = castw(71, G3 * HH);
  h16_t* gWhh0 = castw(48, G3 * HH);
  h16_t* gWih0 = castw(49, G3 * 256);
  h16_t* gWhh1 = castw(52, G3 * HH);
  h16_t* gWih1 = castw(53, G3 * HH);
  h16_t* chW1  = castw(8, HH * 256);
  h16_t* chW2  = castw(9, 64 * HH);
  h16_t* ghW1  = castw(56, HH * 256);
  h16_t* ghW2  = castw(57, 64 * HH);

  auto gemm = [&](const h16_t* A, const h16_t* W, const float* bias,
                  float* oF, h16_t* oH, int M, int Nout, int K, int act) {
    dim3 g(M / 32, (Nout + 127) / 128);
    gemm16_kernel<<<g, 256, 0, stream>>>(A, W, bias, oF, oH, M, Nout, K, act);
  };

  // ---- encoders ----
  lin_small_kernel<<<TN / 2, 256, 0, stream>>>(x_in, 10, P(74), P(76), 1, P(79), P(78), q16d, TN);
  gemm(q16d, nW2, P(77), xf, xf16, TN, HH, HH, 0);                    // xh
  lin_small_kernel<<<TE / 2, 256, 0, stream>>>(ea_in, 6, P(44), P(46), 0, nullptr, nullptr, ebuf, TE);
  gemm(ebuf, eW2, P(47), nullptr, ea16, TE, HH, HH, 0);               // ea
  lin_small_kernel<<<TB / 2, 256, 0, stream>>>(u_in, 5, P(62), P(64), 0, nullptr, nullptr, tmpu, TB);
  gemm(tmpu, gW2, P(65), uhf, uh16, TB, HH, HH, 0);                   // uh

  // ---- 3 transformer-conv rounds ----
  for (int i = 0; i < 3; ++i) {
    int ci = 14 + i * 10;  // be=+5 bk=+6 bq=+7 bs=+8 bv=+9
    gemm(xf16, cW[i][2], P(ci + 7), nullptr, q16d, TN, HH, HH, 0);    // q
    gemm(xf16, cW[i][1], P(ci + 6), nullptr, k16d, TN, HH, HH, 0);    // k
    gemm(xf16, cW[i][4], P(ci + 9), nullptr, v16d, TN, HH, HH, 0);    // v
    gemm(xf16, cW[i][3], P(ci + 8), xi, nullptr, TN, HH, HH, 0);      // skip path
    gemm(ea16, cW[i][0], P(ci + 5), nullptr, ebuf, TE, HH, HH, 0);    // e
    tconv_attn_kernel<<<(TT * BB) / 8, 256, 0, stream>>>(q16d, k16d, v16d, ebuf, xi, TT * BB);
    ln_residual_kernel<<<TN / 2, 256, 0, stream>>>(xf, xi, P(81 + i * 2), P(80 + i * 2), xf, xf16, TN);
  }

  // ---- gate + (T,N)->(N,T) transpose ----
  gate_kernel<<<(TN * HH) / 256, 256, 0, stream>>>(xf, uhf, xseq);

  // ---- node GRU (2 layers) ----
  gemm(xseq, nWih0, P(69), gi, nullptr, TN, G3, HH, 0);
  gru_scan_kernel<<<NN / 16, 256, 0, stream>>>(gi, nWhh0, P(68), ys1, TT);
  gemm(ys1, nWih1, P(73), gi, nullptr, TN, G3, HH, 0);
  gru_scan_kernel<<<NN / 16, 256, 0, stream>>>(gi, nWhh1, P(72), ys2, TT);

  // ---- global GRU (2 layers) over concat(u, node_max) ----
  nodemax_kernel<<<(BB * TT * HH) / 256, 256, 0, stream>>>(ys2, uh16, gin16);
  gemm(gin16, gWih0, P(51), gi, nullptr, TB, G3, 256, 0);
  gru_scan_kernel<<<BB / 16, 256, 0, stream>>>(gi, gWhh0, P(50), gys1, TT);
  gemm(gys1, gWih1, P(55), gi, nullptr, TB, G3, HH, 0);
  gru_scan_kernel<<<BB / 16, 256, 0, stream>>>(gi, gWhh1, P(54), gys2, TT);

  // ---- heads ----
  ncbuild_kernel<<<(NN * HH) / 256, 256, 0, stream>>>(ys2, xseq, nc16);
  gemm(nc16, chW1, P(11), nullptr, ch1, NN, HH, 256, 1);
  gemm(ch1, chW2, P(12), nullptr, ch2, NN, 64, HH, 1);
  contact_final_kernel<<<NN / 256, 256, 0, stream>>>(ch2, P(10), P(13), (float*)d_out);

  attnpool_kernel<<<BB, 128, 0, stream>>>(ys2, P(6), P(7), gys2, gc16);
  gemm(gc16, ghW1, P(59), nullptr, gg1, BB, HH, 256, 1);
  gemm(gg1, ghW2, P(60), nullptr, gg2, BB, 64, HH, 1);
  global_final_kernel<<<(BB * 4 + 255) / 256, 256, 0, stream>>>(gg2, P(58), P(61), (float*)d_out);
}
